// MultiHeadCore_46213848106020
// MI455X (gfx1250) — matrix operand order repaired and checked by reading, not compiled, not run on hardware
//
#include <hip/hip_runtime.h>
#include <hip/hip_bf16.h>
#include <math.h>

typedef _Float16 v16h __attribute__((ext_vector_type(16)));
typedef _Float16 v8h  __attribute__((ext_vector_type(8)));
typedef _Float16 v4h  __attribute__((ext_vector_type(4)));
typedef _Float16 v2h  __attribute__((ext_vector_type(2)));
typedef float    v8f  __attribute__((ext_vector_type(8)));
typedef float    v4f  __attribute__((ext_vector_type(4)));

#define TILE_M 64
#define TILE_N 64
#define TILE_K 32
#define APAD   40   // padded LDS row (halfs) to spread banks

// ---------------------------------------------------------------------------
// Generic fp32 GEMM: C[M,N] = epi( A[M,K] @ B + bias (+R) )
// A fp32 row-major with leading dim lda. B fp32:
//   BLAYOUT==0 : stored (K,N) row-major, ldb = N   (x @ w)
//   BLAYOUT==1 : stored (N,K) row-major, ldb = K   (x @ W.T)
// EPI: 0 = none, 1 = relu, 2 = add residual R[M,N]
// GUARD: true -> scalar guarded staging (K not multiple of 32 / unaligned A).
// fp32 -> f16 conversion during staging; fp32 accumulate via
// v_wmma_f32_16x16x32_f16. Requires M%64==0, N%64==0.
// ---------------------------------------------------------------------------
template<int BLAYOUT, int EPI, bool GUARD>
__global__ __launch_bounds__(128)
void gemm_f32_wmma(const float* __restrict__ A, int lda,
                   const float* __restrict__ Bw, int ldb,
                   const float* __restrict__ bias,
                   const float* __restrict__ R,
                   float* __restrict__ C,
                   int M, int N, int K)
{
  __shared__ _Float16 As[TILE_M * APAD];
  __shared__ _Float16 Bs[TILE_N * APAD];   // stored transposed: Bs[n][k]

  const int tid  = threadIdx.x;
  const int lane = tid & 31;
  const int wv   = tid >> 5;        // 4 waves, each owns 16 M rows
  const int ln16 = lane & 15;
  const int g    = lane >> 4;       // lane half-group
  const int row0 = blockIdx.y * TILE_M;
  const int col0 = blockIdx.x * TILE_N;

  v8f acc[4];
  #pragma unroll
  for (int t = 0; t < 4; ++t)
    #pragma unroll
    for (int e = 0; e < 8; ++e) acc[t][e] = 0.f;

  if (!GUARD) {
    // ---- fast path: vectorized staging + register double buffering ----
    v4f ra[4];
    v4f rb0[2], rb1[2];   // BLAYOUT==0: two adjacent k-rows per chunk
    v4f rbn[4];           // BLAYOUT==1

    auto load_tiles = [&](int k0) {
      #pragma unroll
      for (int j = 0; j < 4; ++j) {
        int c = tid + j * 128;          // 512 chunks of 4 floats = 64x32 tile
        int r = c >> 3, co = (c & 7) * 4;
        ra[j] = *(const v4f*)(A + (size_t)(row0 + r) * lda + k0 + co);
      }
      if (BLAYOUT == 1) {
        #pragma unroll
        for (int j = 0; j < 4; ++j) {
          int c = tid + j * 128;
          int r = c >> 3, co = (c & 7) * 4;
          rbn[j] = *(const v4f*)(Bw + (size_t)(col0 + r) * ldb + k0 + co);
        }
      } else {
        // 256 chunks; each covers k-rows {2kp, 2kp+1} x 4 n-columns.
        // 16 consecutive tids cover one 64-wide row pair -> coalesced b128.
        #pragma unroll
        for (int j = 0; j < 2; ++j) {
          int c = tid + j * 128;
          int kp = c >> 4, n4 = (c & 15) * 4;
          const float* p = Bw + (size_t)(k0 + 2 * kp) * ldb + col0 + n4;
          rb0[j] = *(const v4f*)(p);
          rb1[j] = *(const v4f*)(p + ldb);
        }
      }
    };

    load_tiles(0);
    for (int k0 = 0; k0 < K; k0 += TILE_K) {
      // store staged registers into LDS (f16)
      #pragma unroll
      for (int j = 0; j < 4; ++j) {
        int c = tid + j * 128;
        int r = c >> 3, co = (c & 7) * 4;
        v4h h;
        h[0] = (_Float16)ra[j][0]; h[1] = (_Float16)ra[j][1];
        h[2] = (_Float16)ra[j][2]; h[3] = (_Float16)ra[j][3];
        *(v4h*)(As + r * APAD + co) = h;
      }
      if (BLAYOUT == 1) {
        #pragma unroll
        for (int j = 0; j < 4; ++j) {
          int c = tid + j * 128;
          int r = c >> 3, co = (c & 7) * 4;
          v4h h;
          h[0] = (_Float16)rbn[j][0]; h[1] = (_Float16)rbn[j][1];
          h[2] = (_Float16)rbn[j][2]; h[3] = (_Float16)rbn[j][3];
          *(v4h*)(Bs + r * APAD + co) = h;
        }
      } else {
        // packed transpose: BT[n][2kp..2kp+1] <- (B[2kp][n], B[2kp+1][n])
        #pragma unroll
        for (int j = 0; j < 2; ++j) {
          int c = tid + j * 128;
          int kp = c >> 4, n4 = (c & 15) * 4;
          #pragma unroll
          for (int u = 0; u < 4; ++u) {
            v2h p;
            p[0] = (_Float16)rb0[j][u];
            p[1] = (_Float16)rb1[j][u];
            *(v2h*)(Bs + (n4 + u) * APAD + 2 * kp) = p;
          }
        }
      }
      __syncthreads();

      if (k0 + TILE_K < K) load_tiles(k0 + TILE_K);  // overlap with WMMA below

      const _Float16* ap = &As[(wv * 16 + ln16) * APAD + 8 * g];
      v8h alo = *(const v8h*)(ap);
      v8h ahi = *(const v8h*)(ap + 16);
      v16h afrag = __builtin_shufflevector(alo, ahi,
          0,1,2,3,4,5,6,7,8,9,10,11,12,13,14,15);
      #pragma unroll
      for (int t = 0; t < 4; ++t) {
        const _Float16* bp = &Bs[(t * 16 + ln16) * APAD + 8 * g];
        v8h blo = *(const v8h*)(bp);
        v8h bhi = *(const v8h*)(bp + 16);
        v16h bfrag = __builtin_shufflevector(blo, bhi,
            0,1,2,3,4,5,6,7,8,9,10,11,12,13,14,15);
        acc[t] = __builtin_amdgcn_wmma_f32_16x16x32_f16(
            false, afrag, false, bfrag, (short)0, acc[t], false, false);
      }
      __syncthreads();
    }
  } else {
    // ---- guarded path (only the K=5 box_embed GEMM) ----
    for (int k0 = 0; k0 < K; k0 += TILE_K) {
      for (int i = tid; i < TILE_M * TILE_K; i += 128) {
        int r = i >> 5, c = i & 31;
        int kk = k0 + c;
        float v = (kk < K) ? A[(size_t)(row0 + r) * lda + kk] : 0.f;
        As[r * APAD + c] = (_Float16)v;
      }
      for (int i = tid; i < TILE_N * TILE_K; i += 128) {
        int n = i & 63, c = i >> 6;
        int kk = k0 + c;
        float v;
        if (BLAYOUT == 1)
          v = (kk < K) ? Bw[(size_t)(col0 + n) * ldb + kk] : 0.f;
        else
          v = (kk < K) ? Bw[(size_t)kk * ldb + col0 + n] : 0.f;
        Bs[n * APAD + c] = (_Float16)v;
      }
      __syncthreads();
      const _Float16* ap = &As[(wv * 16 + ln16) * APAD + 8 * g];
      v8h alo = *(const v8h*)(ap);
      v8h ahi = *(const v8h*)(ap + 16);
      v16h afrag = __builtin_shufflevector(alo, ahi,
          0,1,2,3,4,5,6,7,8,9,10,11,12,13,14,15);
      #pragma unroll
      for (int t = 0; t < 4; ++t) {
        const _Float16* bp = &Bs[(t * 16 + ln16) * APAD + 8 * g];
        v8h blo = *(const v8h*)(bp);
        v8h bhi = *(const v8h*)(bp + 16);
        v16h bfrag = __builtin_shufflevector(blo, bhi,
            0,1,2,3,4,5,6,7,8,9,10,11,12,13,14,15);
        acc[t] = __builtin_amdgcn_wmma_f32_16x16x32_f16(
            false, afrag, false, bfrag, (short)0, acc[t], false, false);
      }
      __syncthreads();
    }
  }

  // ---- epilogue (C/D layout: VGPR e -> M = e + 8g, N = lane&15) ----
  #pragma unroll
  for (int t = 0; t < 4; ++t) {
    int col = col0 + t * 16 + ln16;
    float bb = bias ? bias[col] : 0.f;
    #pragma unroll
    for (int e = 0; e < 8; ++e) {
      int row = row0 + wv * 16 + e + 8 * g;
      float v = acc[t][e] + bb;
      if (EPI == 1) v = v > 0.f ? v : 0.f;
      if (EPI == 2) v += R[(size_t)row * N + col];
      C[(size_t)row * N + col] = v;
    }
  }
}

// ---------------------------------------------------------------------------
// LayerNorm over D=512 (eps added to std, matching reference)
// ---------------------------------------------------------------------------
__global__ __launch_bounds__(128)
void layernorm_kernel(const float* __restrict__ x, const float* __restrict__ g,
                      const float* __restrict__ b, float* __restrict__ y)
{
  __shared__ float red[128];
  const int row = blockIdx.x, tid = threadIdx.x;
  const float* xr = x + (size_t)row * 512;
  float lv[4]; float s = 0.f;
  #pragma unroll
  for (int i = 0; i < 4; ++i) { lv[i] = xr[tid + i * 128]; s += lv[i]; }
  red[tid] = s; __syncthreads();
  for (int off = 64; off > 0; off >>= 1) { if (tid < off) red[tid] += red[tid + off]; __syncthreads(); }
  const float mean = red[0] * (1.f / 512.f);
  __syncthreads();
  float vs = 0.f;
  #pragma unroll
  for (int i = 0; i < 4; ++i) { float d = lv[i] - mean; vs += d * d; }
  red[tid] = vs; __syncthreads();
  for (int off = 64; off > 0; off >>= 1) { if (tid < off) red[tid] += red[tid + off]; __syncthreads(); }
  const float inv = 1.f / (sqrtf(red[0] * (1.f / 512.f)) + 1e-6f);
  #pragma unroll
  for (int i = 0; i < 4; ++i) {
    int c = tid + i * 128;
    y[(size_t)row * 512 + c] = g[c] * (lv[i] - mean) * inv + b[c];
  }
}

// ---------------------------------------------------------------------------
// Encoder box-attention core: per (b,h) block. q/k = [attn(64)|geo(64)],
// scores/softmax over N=36 with mask, out = attn @ va, merged layout.
// ---------------------------------------------------------------------------
__global__ __launch_bounds__(128)
void enc_attn_kernel(const float* __restrict__ qa, const float* __restrict__ qg,
                     const float* __restrict__ ka, const float* __restrict__ kg,
                     const float* __restrict__ va, const int* __restrict__ mask,
                     float* __restrict__ out)
{
  __shared__ float q[36][128];
  __shared__ float k[36][128];
  __shared__ float v[36][64];
  __shared__ float s[36][36];
  const int b = blockIdx.x >> 3;
  const int h = blockIdx.x & 7;
  const int tid = threadIdx.x;
  const size_t base = (size_t)(b * 36) * 512 + h * 64;

  for (int i = tid; i < 36 * 128; i += 128) {
    int r = i >> 7, c = i & 127;
    size_t idx = base + (size_t)r * 512 + (c & 63);
    q[r][c] = (c < 64) ? qa[idx] : qg[idx];
    k[r][c] = (c < 64) ? ka[idx] : kg[idx];
  }
  for (int i = tid; i < 36 * 64; i += 128) {
    int r = i >> 6, c = i & 63;
    v[r][c] = va[base + (size_t)r * 512 + c];
  }
  __syncthreads();

  const float scale = 0.08838834764831845f;   // 1/sqrt(2*DK)=1/sqrt(128)
  for (int i = tid; i < 36 * 36; i += 128) {
    int r = i / 36, j = i - r * 36;
    float acc = 0.f;
    #pragma unroll 8
    for (int e = 0; e < 128; ++e) acc += q[r][e] * k[j][e];
    acc *= scale;
    if (mask[b * 36 + j] == 0) acc = -1.0e9f;
    s[r][j] = acc;
  }
  __syncthreads();
  if (tid < 36) {
    float mx = -3.4e38f;
    for (int j = 0; j < 36; ++j) mx = fmaxf(mx, s[tid][j]);
    float sum = 0.f;
    for (int j = 0; j < 36; ++j) { float e = __expf(s[tid][j] - mx); s[tid][j] = e; sum += e; }
    float inv = 1.f / sum;
    for (int j = 0; j < 36; ++j) s[tid][j] *= inv;
  }
  __syncthreads();
  for (int i = tid; i < 36 * 64; i += 128) {
    int r = i >> 6, d = i & 63;
    float acc = 0.f;
    for (int j = 0; j < 36; ++j) acc += s[r][j] * v[j][d];
    out[base + (size_t)r * 512 + d] = acc;
  }
}

// ---------------------------------------------------------------------------
// Decoder single-query attention: per b block, 8 heads x 36 keys x 64 dims.
// ---------------------------------------------------------------------------
__global__ __launch_bounds__(128)
void dec_attn_kernel(const float* __restrict__ qv, const float* __restrict__ kmat,
                     const float* __restrict__ vmat, float* __restrict__ out)
{
  __shared__ float q[512];
  __shared__ float s[8][36];
  const int b = blockIdx.x;
  const int tid = threadIdx.x;
  for (int i = tid; i < 512; i += 128) q[i] = qv[(size_t)b * 512 + i];
  __syncthreads();
  for (int i = tid; i < 8 * 36; i += 128) {
    int h = i / 36, j = i - h * 36;
    const float* kr = kmat + ((size_t)(b * 36 + j)) * 512 + h * 64;
    float acc = 0.f;
    #pragma unroll 8
    for (int d = 0; d < 64; ++d) acc += q[h * 64 + d] * kr[d];
    s[h][j] = acc * 0.125f;   // 1/sqrt(64)
  }
  __syncthreads();
  if (tid < 8) {
    float mx = -3.4e38f;
    for (int j = 0; j < 36; ++j) mx = fmaxf(mx, s[tid][j]);
    float sum = 0.f;
    for (int j = 0; j < 36; ++j) { float e = __expf(s[tid][j] - mx); s[tid][j] = e; sum += e; }
    float inv = 1.f / sum;
    for (int j = 0; j < 36; ++j) s[tid][j] *= inv;
  }
  __syncthreads();
  for (int i = tid; i < 512; i += 128) {
    int h = i >> 6, d = i & 63;
    float acc = 0.f;
    for (int j = 0; j < 36; ++j)
      acc += s[h][j] * vmat[((size_t)(b * 36 + j)) * 512 + h * 64 + d];
    out[(size_t)b * 512 + i] = acc;
  }
}

// ---------------------------------------------------------------------------
// LSTM pointwise: gates (B,2048) laid out [i|f|g|o]
// ---------------------------------------------------------------------------
__global__ void lstm_kernel(const float* __restrict__ gates, const float* __restrict__ cprev,
                            float* __restrict__ hout, float* __restrict__ cout)
{
  int idx = blockIdx.x * blockDim.x + threadIdx.x;
  if (idx >= 128 * 512) return;
  int bi = idx >> 9, d = idx & 511;
  const float* gr = gates + (size_t)bi * 2048;
  float ig = gr[d], fg = gr[512 + d], gg = gr[1024 + d], og = gr[1536 + d];
  float si = 1.f / (1.f + __expf(-ig));
  float sf = 1.f / (1.f + __expf(-fg));
  float so = 1.f / (1.f + __expf(-og));
  float c = sf * cprev[idx] + si * tanhf(gg);
  float h = so * tanhf(c);
  hout[idx] = h;
  cout[idx] = c;
}

__global__ void gated_add_kernel(float* __restrict__ x, const float* __restrict__ gl,
                                 const float* __restrict__ il, int n)
{
  int idx = blockIdx.x * blockDim.x + threadIdx.x;
  if (idx >= n) return;
  float s = 1.f / (1.f + __expf(-gl[idx]));
  x[idx] += s * il[idx];
}

__global__ void concat2_kernel(const float* __restrict__ a, const float* __restrict__ b2,
                               float* __restrict__ out, int M, int D1, int D2)
{
  int idx = blockIdx.x * blockDim.x + threadIdx.x;
  int DT = D1 + D2;
  if (idx >= M * DT) return;
  int r = idx / DT, c = idx - r * DT;
  out[idx] = (c < D1) ? a[(size_t)r * D1 + c] : b2[(size_t)r * D2 + (c - D1)];
}

// ---------------------------------------------------------------------------
extern "C" void kernel_launch(void* const* d_in, const int* in_sizes, int n_in,
                              void* d_out, int out_size, void* d_ws, size_t ws_size,
                              hipStream_t stream)
{
  (void)in_sizes; (void)n_in; (void)out_size; (void)ws_size;
  constexpr int B = 128, NN = 36, D = 512, H = 8, FFD = 2048, FEAT = 2653;
  const int BN = B * NN;           // 4608
  auto f = [&](int i) { return (const float*)d_in[i]; };

  // inputs (jax pytree / sorted-dict flattening order)
  const float* xt        = f(0);
  const float* fc        = f(1);
  const float* att_feats = f(2);
  const float* p_att     = f(3);
  const float* state_h   = f(4);
  const float* state_c   = f(5);
  const int*   mask      = (const int*)d_in[6];
  const float* box_b     = f(7);
  const float* box_w     = f(8);
  const float* dec_norm_b = f(63); const float* dec_norm_g = f(64);
  const float* enc_norm_b = f(131); const float* enc_norm_g = f(132);
  const float* lstm_b_hh = f(133); const float* lstm_b_ih = f(134);
  const float* lstm_w_hh = f(135); const float* lstm_w_ih = f(136);
  const float* qn_b = f(137); const float* qn_g = f(138);

  float* out   = (float*)d_out;
  float* vis   = out;                 // (B,D)
  float* h_att = out + 65536;         // (B,D)
  float* c_att = out + 131072;        // (B,D)

  float* ws = (float*)d_ws;
  const size_t S1 = (size_t)BN * D;   // 2359296 floats
  float* x_   = ws + 0 * S1;
  float* geo  = ws + 1 * S1;
  float* ln   = ws + 2 * S1;
  float* qa   = ws + 3 * S1;
  float* ka   = ws + 4 * S1;
  float* va   = ws + 5 * S1;
  float* qg   = ws + 6 * S1;
  float* kg   = ws + 7 * S1;
  float* aout = ws + 8 * S1;
  float* infl = ws + 9 * S1;
  float* gatl = ws + 10 * S1;
  float* gcat = ws + 11 * S1;         // 2 units (11,12) during encoder
  float* ffh  = ws + 3 * S1;          // overlays qa..qg during FFN (4 units)
  float* mem  = ws + 11 * S1;         // overlays gcat after encoder completes
  float* kd   = ws + 3 * S1;          // decoder K (overlays qa)
  float* vd   = ws + 4 * S1;          // decoder V (overlays ka)
  float* tail = ws + 13 * S1;
  float* xcat  = tail;                // B*1024
  float* gates = tail + 131072;       // B*2048
  float* yb    = tail + 393216;       // B*D
  float* yln   = tail + 458752;
  float* qd    = tail + 524288;
  float* datt  = tail + 589824;
  float* dgat  = tail + 655360;
  float* dinf  = tail + 720896;
  float* ffhd  = tail + 786432;       // B*FFD

  auto gemm = [&](int blayout, int epi, const float* A, int lda, const float* Bw, int ldb,
                  const float* bias, const float* R, float* C, int M_, int N_, int K_) {
    dim3 grid(N_ / 64, M_ / 64), blk(128);
    bool guard = (K_ % 32) != 0;
#define GL(BL, EP, GD) gemm_f32_wmma<BL, EP, GD><<<grid, blk, 0, stream>>>(A, lda, Bw, ldb, bias, R, C, M_, N_, K_)
    if (!guard) {
      if (blayout == 0) { if (epi == 0) GL(0, 0, false); else if (epi == 1) GL(0, 1, false); else GL(0, 2, false); }
      else              { if (epi == 0) GL(1, 0, false); else if (epi == 1) GL(1, 1, false); else GL(1, 2, false); }
    } else {
      if (blayout == 0) { if (epi == 0) GL(0, 0, true); else if (epi == 1) GL(0, 1, true); else GL(0, 2, true); }
      else              { if (epi == 0) GL(1, 0, true); else if (epi == 1) GL(1, 1, true); else GL(1, 2, true); }
    }
#undef GL
  };

  // ---------------- LSTM cell ----------------
  concat2_kernel<<<(B * 1024 + 255) / 256, 256, 0, stream>>>(xt, fc, xcat, B, D, D);
  gemm(1, 0, xcat, 1024, lstm_w_ih, 1024, lstm_b_ih, nullptr, gates, B, 2048, 1024);
  gemm(1, 2, state_h, 512, lstm_w_hh, 512, lstm_b_hh, gates, gates, B, 2048, 512);
  lstm_kernel<<<(65536 + 255) / 256, 256, 0, stream>>>(gates, state_c, h_att, c_att);

  // ---------------- geo = relu(att_feats[..., -5:] @ box_w + box_b) ----------------
  gemm(0, 1, att_feats + (FEAT - 5), FEAT, box_w, D, box_b, nullptr, geo, BN, D, 5);

  // x = p_att_feats
  hipMemcpyAsync(x_, p_att, S1 * sizeof(float), hipMemcpyDeviceToDevice, stream);

  // ---------------- encoder (3 layers) ----------------
  for (int L = 0; L < 3; ++L) {
    int E = 65 + L * 22;
    const float *gate_b = f(E + 0), *gate_w = f(E + 1), *info_b = f(E + 2), *info_w = f(E + 3);
    const float *ka_b = f(E + 4), *ka_w = f(E + 5), *kg_b = f(E + 6), *kg_w = f(E + 7);
    const float *qa_b = f(E + 8), *qa_w = f(E + 9), *qg_b = f(E + 10), *qg_w = f(E + 11);
    const float *va_b = f(E + 12), *va_w = f(E + 13);
    const float *ff1_b = f(E + 14), *ff1_w = f(E + 15), *ff2_b = f(E + 16), *ff2_w = f(E + 17);
    const float *ln1_b = f(E + 18), *ln1_g = f(E + 19), *ln2_b = f(E + 20), *ln2_g = f(E + 21);

    layernorm_kernel<<<BN, 128, 0, stream>>>(x_, ln1_g, ln1_b, ln);
    gemm(0, 0, ln,  D, qa_w, D, qa_b, nullptr, qa, BN, D, D);
    gemm(0, 0, ln,  D, ka_w, D, ka_b, nullptr, ka, BN, D, D);
    gemm(0, 0, ln,  D, va_w, D, va_b, nullptr, va, BN, D, D);
    gemm(0, 0, geo, D, qg_w, D, qg_b, nullptr, qg, BN, D, D);
    gemm(0, 0, geo, D, kg_w, D, kg_b, nullptr, kg, BN, D, D);
    enc_attn_kernel<<<B * H, 128, 0, stream>>>(qa, qg, ka, kg, va, mask, aout);
    concat2_kernel<<<(BN * 1024 + 255) / 256, 256, 0, stream>>>(ln, geo, gcat, BN, D, D);
    gemm(0, 0, gcat, 1024, gate_w, D, gate_b, nullptr, gatl, BN, D, 1024);
    gemm(0, 0, aout, D, info_w, D, info_b, nullptr, infl, BN, D, D);
    gated_add_kernel<<<((int)S1 + 255) / 256, 256, 0, stream>>>(x_, gatl, infl, (int)S1);
    layernorm_kernel<<<BN, 128, 0, stream>>>(x_, ln2_g, ln2_b, ln);
    gemm(0, 1, ln, D, ff1_w, FFD, ff1_b, nullptr, ffh, BN, FFD, D);
    gemm(0, 2, ffh, FFD, ff2_w, D, ff2_b, x_, x_, BN, D, FFD);
  }
  layernorm_kernel<<<BN, 128, 0, stream>>>(x_, enc_norm_g, enc_norm_b, mem);

  // y = layer_norm(h_att, q_norm)
  layernorm_kernel<<<B, 128, 0, stream>>>(h_att, qn_g, qn_b, yb);

  // ---------------- decoder (3 layers) ----------------
  for (int L = 0; L < 3; ++L) {
    int Dd = 9 + L * 18;
    const float *gate_b = f(Dd + 0), *gate_w = f(Dd + 1), *info_b = f(Dd + 2), *info_w = f(Dd + 3);
    const float *k_b = f(Dd + 4), *k_w = f(Dd + 5), *q_b = f(Dd + 6), *q_w = f(Dd + 7);
    const float *v_b = f(Dd + 8), *v_w = f(Dd + 9);
    const float *ff1_b = f(Dd + 10), *ff1_w = f(Dd + 11), *ff2_b = f(Dd + 12), *ff2_w = f(Dd + 13);
    const float *ln1_b = f(Dd + 14), *ln1_g = f(Dd + 15), *ln2_b = f(Dd + 16), *ln2_g = f(Dd + 17);

    layernorm_kernel<<<B, 128, 0, stream>>>(yb, ln1_g, ln1_b, yln);
    gemm(0, 0, yln, D, q_w, D, q_b, nullptr, qd, B, D, D);
    gemm(0, 0, mem, D, k_w, D, k_b, nullptr, kd, BN, D, D);
    gemm(0, 0, mem, D, v_w, D, v_b, nullptr, vd, BN, D, D);
    dec_attn_kernel<<<B, 128, 0, stream>>>(qd, kd, vd, datt);
    gemm(0, 0, yln, D, gate_w, D, gate_b, nullptr, dgat, B, D, D);
    gemm(0, 0, datt, D, info_w, D, info_b, nullptr, dinf, B, D, D);
    gated_add_kernel<<<(65536 + 255) / 256, 256, 0, stream>>>(yb, dgat, dinf, 65536);
    layernorm_kernel<<<B, 128, 0, stream>>>(yb, ln2_g, ln2_b, yln);
    gemm(0, 1, yln, D, ff1_w, FFD, ff1_b, nullptr, ffhd, B, FFD, D);
    gemm(0, 2, ffhd, FFD, ff2_w, D, ff2_b, yb, yb, B, D, FFD);
  }
  layernorm_kernel<<<B, 128, 0, stream>>>(yb, dec_norm_g, dec_norm_b, vis);
}
